// MambaBlock_67448166416409
// MI455X (gfx1250) — compile-verified, matrix operand order checked by reading, not run on hardware
//
#include <hip/hip_runtime.h>
#include <hip/hip_bf16.h>

// ---------------------------------------------------------------------------
// Mamba block for MI455X (gfx1250, wave32).
//  - GEMMs on the WMMA pipe: v_wmma_f32_16x16x32_f16, f16 in / f32 accum.
//  - Selective scan: 1 workgroup per batch, state in registers, per-step
//    rows staged in LDS, wave32 shfl_xor reduction.
// ---------------------------------------------------------------------------

typedef __attribute__((ext_vector_type(16))) _Float16 v16h;
typedef __attribute__((ext_vector_type(8)))  _Float16 v8h;
typedef __attribute__((ext_vector_type(8)))  float    v8f;

#define BATCH   4
#define SEQ     1024
#define M_TOT   (BATCH * SEQ)     // 4096 rows of activations
#define D_MODEL 64
#define D_INNER 128
#define D_STATE 128
#define DT_RANK 4
#define XDBL_LD 272               // 260 padded to 17 WMMA N-tiles

__device__ __forceinline__ float silu_f(float x) {
    return x / (1.0f + __expf(-x));
}
__device__ __forceinline__ float softplus_f(float x) {
    return (x > 20.0f) ? x : __logf(1.0f + __expf(x));
}

// ---------------------------------------------------------------------------
// f32 -> f16 flat conversion
// ---------------------------------------------------------------------------
__global__ void mamba_cvt_f16(const float* __restrict__ in,
                              _Float16* __restrict__ out, int n) {
    int i = blockIdx.x * blockDim.x + threadIdx.x;
    if (i < n) out[i] = (_Float16)in[i];
}

// f32 (rows_in x k) -> f16 (rows_out x k), zero-padding extra rows
__global__ void mamba_pad_f16(const float* __restrict__ in,
                              _Float16* __restrict__ out,
                              int rows_in, int rows_out, int k) {
    int i = blockIdx.x * blockDim.x + threadIdx.x;
    if (i < rows_out * k) {
        int r = i / k;
        out[i] = (r < rows_in) ? (_Float16)in[i] : (_Float16)0.0f;
    }
}

// ---------------------------------------------------------------------------
// WMMA GEMM: C[M x Nld] = A[M x K] * Bt[N x K]^T   (A,Bt f16 row-major)
// One wave (32 lanes) per 16x16 output tile. grid = (Nld/16, M/16).
// A-fragment layout per ISA 7.12.2 (16-bit A 16x32):
//   lanes 0-15 : M=lane,    K = k0+{0..7, 16..23}
//   lanes 16-31: M=lane-16, K = k0+{8..15, 24..31}
// B-fragment: lane holds column N, same K striping (Bt row is contiguous).
// ---------------------------------------------------------------------------
template <bool BIAS>
__global__ void mamba_wmma_gemm(const _Float16* __restrict__ A,
                                const _Float16* __restrict__ Bt,
                                float* __restrict__ C,
                                const float* __restrict__ bias,
                                int K, int Nld) {
    const int lane = threadIdx.x;          // 0..31
    const int n0   = blockIdx.x * 16;
    const int m0   = blockIdx.y * 16;
    const int hsel = lane >> 4;            // 0 or 1
    const int sub  = lane & 15;

    const _Float16* arow = A  + (size_t)(m0 + sub) * K + hsel * 8;
    const _Float16* brow = Bt + (size_t)(n0 + sub) * K + hsel * 8;

    v8f acc = {};
    for (int k0 = 0; k0 < K; k0 += 32) {
        v8h alo = *(const v8h*)(arow + k0);
        v8h ahi = *(const v8h*)(arow + k0 + 16);
        v8h blo = *(const v8h*)(brow + k0);
        v8h bhi = *(const v8h*)(brow + k0 + 16);
        v16h a, b;
#pragma unroll
        for (int i = 0; i < 8; ++i) {
            a[i] = alo[i]; a[8 + i] = ahi[i];
            b[i] = blo[i]; b[8 + i] = bhi[i];
        }
        acc = __builtin_amdgcn_wmma_f32_16x16x32_f16(
            /*neg_a=*/false, a, /*neg_b=*/false, b,
            /*c_mod=*/(short)0, acc, /*reuse_a=*/false, /*reuse_b=*/false);
    }

    // C/D layout: VGPR r -> row m0 + r + hsel*8, col n0 + sub
    const int col   = n0 + sub;
    const int rbase = m0 + hsel * 8;
    float badd = 0.0f;
    if (BIAS) badd = bias[col];
#pragma unroll
    for (int r = 0; r < 8; ++r) {
        C[(size_t)(rbase + r) * Nld + col] = acc[r] + badd;
    }
}

// ---------------------------------------------------------------------------
// Causal depthwise conv (K=4) + SiLU.  Reads xi half of xz (cols 0..127),
// writes u (f32, scan input) and xi16 (f16, x_proj GEMM input).
// ---------------------------------------------------------------------------
__global__ void mamba_conv_silu(const float* __restrict__ xz,
                                const float* __restrict__ cw,
                                const float* __restrict__ cb,
                                float* __restrict__ u,
                                _Float16* __restrict__ xi16) {
    int i = blockIdx.x * blockDim.x + threadIdx.x;
    if (i >= M_TOT * D_INNER) return;
    int d = i & (D_INNER - 1);
    int m = i >> 7;                 // row index (b*SEQ + t)
    int t = m & (SEQ - 1);
    int base = m - t;
    float acc = cb[d];
#pragma unroll
    for (int k = 0; k < 4; ++k) {
        int tt = t - 3 + k;
        if (tt >= 0)
            acc += xz[(size_t)(base + tt) * 256 + d] * cw[d * 4 + k];
    }
    float y = silu_f(acc);
    u[i] = y;
    xi16[i] = (_Float16)y;
}

// ---------------------------------------------------------------------------
// delta = softplus(xdbl[:, :4] @ dt_w^T + dt_b)
// ---------------------------------------------------------------------------
__global__ void mamba_delta(const float* __restrict__ xdbl,
                            const float* __restrict__ dtw,
                            const float* __restrict__ dtb,
                            float* __restrict__ delta) {
    int i = blockIdx.x * blockDim.x + threadIdx.x;
    if (i >= M_TOT * D_INNER) return;
    int d = i & (D_INNER - 1);
    int m = i >> 7;
    const float* xr = xdbl + (size_t)m * XDBL_LD;
    float acc = dtb[d];
#pragma unroll
    for (int r = 0; r < DT_RANK; ++r)
        acc += xr[r] * dtw[d * DT_RANK + r];
    delta[i] = softplus_f(acc);
}

// ---------------------------------------------------------------------------
// Selective scan. grid = BATCH, block = 1024 threads.
// thread -> (d = tid/8, n-slice = (tid&7)*16 .. +15); 16 states in registers.
// Per step: stage delta/u/B/C rows in LDS, recurrence + 3x shfl_xor reduce.
// ---------------------------------------------------------------------------
__global__ void __launch_bounds__(1024)
mamba_scan(const float* __restrict__ u,
           const float* __restrict__ delta,
           const float* __restrict__ xdbl,   // B at +4, C at +132, ld=272
           const float* __restrict__ A_log,
           const float* __restrict__ Dp,
           float* __restrict__ y) {
    const int tid = threadIdx.x;
    const int b   = blockIdx.x;
    const int d   = tid >> 3;
    const int j   = tid & 7;
    const int n0  = j * 16;

    __shared__ float sB[D_STATE];
    __shared__ float sC[D_STATE];
    __shared__ float sDU[2 * D_INNER];

    float acoef[16];
#pragma unroll
    for (int i = 0; i < 16; ++i)
        acoef[i] = -__expf(A_log[d * D_STATE + n0 + i]);
    const float dcoef = Dp[d];

    float h[16];
#pragma unroll
    for (int i = 0; i < 16; ++i) h[i] = 0.0f;

    for (int t = 0; t < SEQ; ++t) {
        const int m = b * SEQ + t;
        __syncthreads();               // protect LDS from previous step
        if (tid < 128)       sDU[tid]        = delta[(size_t)m * D_INNER + tid];
        else if (tid < 256)  sDU[tid]        = u[(size_t)m * D_INNER + (tid - 128)];
        else if (tid < 384)  sB[tid - 256]   = xdbl[(size_t)m * XDBL_LD + 4 + (tid - 256)];
        else if (tid < 512)  sC[tid - 384]   = xdbl[(size_t)m * XDBL_LD + 132 + (tid - 384)];
        __syncthreads();

        const float dt = sDU[d];
        const float uu = sDU[128 + d];
        const float dtu = dt * uu;
        float partial = 0.0f;
#pragma unroll
        for (int i = 0; i < 16; ++i) {
            float dA = __expf(dt * acoef[i]);
            h[i] = dA * h[i] + dtu * sB[n0 + i];
            partial += h[i] * sC[n0 + i];
        }
        // reduce the 8 lanes sharing channel d (contiguous lanes in a wave32)
        partial += __shfl_xor(partial, 1);
        partial += __shfl_xor(partial, 2);
        partial += __shfl_xor(partial, 4);
        if (j == 0)
            y[(size_t)m * D_INNER + d] = partial + dcoef * uu;
    }
}

// ---------------------------------------------------------------------------
// gated = y * silu(z);  write f16 for out_proj GEMM.  z = xz cols 128..255.
// ---------------------------------------------------------------------------
__global__ void mamba_gate(const float* __restrict__ y,
                           const float* __restrict__ xz,
                           _Float16* __restrict__ g16) {
    int i = blockIdx.x * blockDim.x + threadIdx.x;
    if (i >= M_TOT * D_INNER) return;
    int d = i & (D_INNER - 1);
    int m = i >> 7;
    float z = xz[(size_t)m * 256 + 128 + d];
    g16[i] = (_Float16)(y[i] * silu_f(z));
}

// ---------------------------------------------------------------------------
// host side
// ---------------------------------------------------------------------------
extern "C" void kernel_launch(void* const* d_in, const int* in_sizes, int n_in,
                              void* d_out, int out_size, void* d_ws, size_t ws_size,
                              hipStream_t stream) {
    (void)in_sizes; (void)n_in; (void)out_size; (void)ws_size;

    const float* x      = (const float*)d_in[0];
    const float* in_w   = (const float*)d_in[1];   // (2,256,64)
    const float* conv_w = (const float*)d_in[2];   // (2,128,4)
    const float* conv_b = (const float*)d_in[3];   // (2,128)
    const float* xp_w   = (const float*)d_in[4];   // (2,260,128)
    const float* dt_w   = (const float*)d_in[5];   // (2,128,4)
    const float* dt_b   = (const float*)d_in[6];   // (2,128)
    const float* A_log  = (const float*)d_in[7];   // (2,128,128)
    const float* Dvec   = (const float*)d_in[8];   // (2,128)
    const float* out_w  = (const float*)d_in[9];   // (2,64,128)
    const float* fc_w   = (const float*)d_in[10];  // (64,64)
    const float* fc_b   = (const float*)d_in[11];  // (64,)

    char* ws = (char*)d_ws;                         // all offsets 256B aligned
    float*    xz    = (float*)   (ws + 0);          // 4096x256 f32
    float*    u     = (float*)   (ws + 4194304);    // 4096x128 f32
    float*    xdbl  = (float*)   (ws + 6291456);    // 4096x272 f32
    float*    delta = (float*)   (ws + 10747904);   // 4096x128 f32
    float*    ysc   = (float*)   (ws + 12845056);   // 4096x128 f32
    float*    hbuf  = (float*)   (ws + 14942208);   // 4096x64  f32
    _Float16* a16   = (_Float16*)(ws + 15990784);   // 4096x128 f16 (reused)
    _Float16* w_in  = (_Float16*)(ws + 17039360);   // 256x64
    _Float16* w_xp  = (_Float16*)(ws + 17072128);   // 272x128
    _Float16* w_op  = (_Float16*)(ws + 17141760);   // 64x128
    _Float16* w_fc  = (_Float16*)(ws + 17158144);   // 64x64

    const int T = 256;
    const int gMD = (M_TOT * D_INNER + T - 1) / T;  // 4096*128 threads

    for (int l = 0; l < 2; ++l) {
        const float* hin = (l == 0) ? x : hbuf;

        // in_proj: xz = h @ in_w^T   (M=4096, N=256, K=64)
        mamba_cvt_f16<<<(M_TOT * D_MODEL + T - 1) / T, T, 0, stream>>>(hin, a16, M_TOT * D_MODEL);
        mamba_cvt_f16<<<(256 * 64 + T - 1) / T, T, 0, stream>>>(in_w + (size_t)l * 256 * 64, w_in, 256 * 64);
        mamba_wmma_gemm<false><<<dim3(16, M_TOT / 16), 32, 0, stream>>>(a16, w_in, xz, nullptr, 64, 256);

        // causal dwconv + silu
        mamba_conv_silu<<<gMD, T, 0, stream>>>(xz, conv_w + (size_t)l * 512,
                                               conv_b + (size_t)l * 128, u, a16);

        // x_proj: xdbl = xi @ xp_w^T  (N=260 padded to 272, K=128)
        mamba_pad_f16<<<(272 * 128 + T - 1) / T, T, 0, stream>>>(xp_w + (size_t)l * 260 * 128, w_xp, 260, 272, 128);
        mamba_wmma_gemm<false><<<dim3(17, M_TOT / 16), 32, 0, stream>>>(a16, w_xp, xdbl, nullptr, 128, XDBL_LD);

        // delta
        mamba_delta<<<gMD, T, 0, stream>>>(xdbl, dt_w + (size_t)l * 512,
                                           dt_b + (size_t)l * 128, delta);

        // selective scan
        mamba_scan<<<BATCH, 1024, 0, stream>>>(u, delta, xdbl,
                                               A_log + (size_t)l * 128 * 128,
                                               Dvec + (size_t)l * 128, ysc);

        // gate, out_proj: hbuf = (y*silu(z)) @ out_w^T  (N=64, K=128)
        mamba_gate<<<gMD, T, 0, stream>>>(ysc, xz, a16);
        mamba_cvt_f16<<<(64 * 128 + T - 1) / T, T, 0, stream>>>(out_w + (size_t)l * 64 * 128, w_op, 64 * 128);
        mamba_wmma_gemm<false><<<dim3(4, M_TOT / 16), 32, 0, stream>>>(a16, w_op, hbuf, nullptr, 128, 64);
    }

    // final fc: out = h @ fc_w^T + fc_b  (N=64, K=64)
    mamba_cvt_f16<<<(M_TOT * D_MODEL + T - 1) / T, T, 0, stream>>>(hbuf, a16, M_TOT * D_MODEL);
    mamba_cvt_f16<<<(64 * 64 + T - 1) / T, T, 0, stream>>>(fc_w, w_fc, 64 * 64);
    mamba_wmma_gemm<true><<<dim3(4, M_TOT / 16), 32, 0, stream>>>(a16, w_fc, (float*)d_out, fc_b, 64, 64);
}
